// LSTMTwin_83992380441307
// MI455X (gfx1250) — matrix operand order repaired and checked by reading, not compiled, not run on hardware
//
#include <hip/hip_runtime.h>
#include <hip/hip_fp16.h>
#include <math.h>

typedef __attribute__((ext_vector_type(16))) _Float16 v16h;
typedef __attribute__((ext_vector_type(8)))  _Float16 v8h;
typedef __attribute__((ext_vector_type(8)))  float    v8f;

namespace {

constexpr int kT = 20;
constexpr int kWStride = 72;                               // padded f16 row stride
constexpr int kRowsPerWave = 16;
constexpr int kWaves = 8;
constexpr int kRowsPerBlock = kRowsPerWave * kWaves;       // 128

// ---- LDS layout (bytes) ----
constexpr int OFF_WH0 = 0;                                 // 256*72*2
constexpr int OFF_WI1 = OFF_WH0 + 256 * kWStride * 2;
constexpr int OFF_WH1 = OFF_WI1 + 256 * kWStride * 2;
constexpr int OFF_WHD = OFF_WH1 + 256 * kWStride * 2;      // 160*72*2
constexpr int OFF_B0  = OFF_WHD + 160 * kWStride * 2;
constexpr int OFF_B1  = OFF_B0 + 256 * 4;
constexpr int OFF_BH  = OFF_B1 + 256 * 4;
constexpr int OFF_WI0 = OFF_BH + 160 * 4;
constexpr int OFF_X   = OFF_WI0 + 512 * 4;
constexpr int OFF_HB  = OFF_X + kWaves * kRowsPerWave * 40 * 4;
constexpr int SMEM_TOTAL = OFF_HB + kWaves * kRowsPerWave * kWStride * 2; // 177280

__device__ __forceinline__ float fast_tanh(float x) {
#if __has_builtin(__builtin_amdgcn_tanhf)
  return __builtin_amdgcn_tanhf(x);        // v_tanh_f32 on gfx1250
#else
  return tanhf(x);
#endif
}
__device__ __forceinline__ float fast_sigmoid(float x) {
  return fmaf(0.5f, fast_tanh(0.5f * x), 0.5f);
}

__device__ __forceinline__ v16h hcat(v8h lo, v8h hi) {
  return __builtin_shufflevector(lo, hi, 0,1,2,3,4,5,6,7,8,9,10,11,12,13,14,15);
}

__device__ __forceinline__ v8f wmma16(v16h a, v16h b, v8f c) {
  // D = A(16x32 f16) * B(32x16 f16) + C(16x16 f32)
  return __builtin_amdgcn_wmma_f32_16x16x32_f16(false, a, false, b, (short)0, c,
                                                false, false);
}

// B fragment (32x16 f16), N-tile nt, K-chunk kc. LDS row r (stride 72, 64 valid)
// holds column r of the B operand (== row r of the original weight matrix).
// Lane layout: lane%16 = N column; lanes 0-15 hold K=kc*32..+15, lanes 16-31 K=+16..+31.
__device__ __forceinline__ v16h load_bfrag(const _Float16* w, int nt, int kc,
                                           int n, int g) {
  const _Float16* p = w + (nt * 16 + n) * kWStride + kc * 32 + g * 16;
  return hcat(*(const v8h*)p, *(const v8h*)(p + 8));
}

// A fragment (16x32 f16), K-chunk kc, from per-wave h buffer (16 rows x 64, stride 72).
// Lane m holds: lanes 0-15 -> K = kc*32 + 0..15; lanes 16-31 -> K = kc*32 + 16..31.
__device__ __forceinline__ v16h load_afrag(const _Float16* hb, int kc, int m, int g) {
  const _Float16* p = hb + m * kWStride + kc * 32 + g * 16;
  return hcat(*(const v8h*)p, *(const v8h*)(p + 8));
}

} // namespace

__global__ __launch_bounds__(256)
void lstm2_head_wmma_kernel(const float* __restrict__ x,
                            const float* __restrict__ Wih0, const float* __restrict__ Whh0,
                            const float* __restrict__ bih0, const float* __restrict__ bhh0,
                            const float* __restrict__ Wih1, const float* __restrict__ Whh1,
                            const float* __restrict__ bih1, const float* __restrict__ bhh1,
                            const float* __restrict__ headw, const float* __restrict__ headb,
                            float* __restrict__ out) {
  extern __shared__ char smem[];
  _Float16* wh0 = (_Float16*)(smem + OFF_WH0);
  _Float16* wi1 = (_Float16*)(smem + OFF_WI1);
  _Float16* wh1 = (_Float16*)(smem + OFF_WH1);
  _Float16* whd = (_Float16*)(smem + OFF_WHD);
  float*    b0f = (float*)(smem + OFF_B0);
  float*    b1f = (float*)(smem + OFF_B1);
  float*    bhf = (float*)(smem + OFF_BH);
  float*    wi0 = (float*)(smem + OFF_WI0);

  const int tid  = threadIdx.x;
  const int wave = tid >> 5;
  const int lane = tid & 31;
  const int g    = lane >> 4;   // half-wave id
  const int n    = lane & 15;   // N column (B/C/D) or M row (A)

  // ---- stage weights (fp32 -> f16, padded rows) ----
  for (int i = tid; i < 256 * 64; i += 256) {
    const int r = i >> 6, k = i & 63;
    wh0[r * kWStride + k] = (_Float16)Whh0[i];
    wi1[r * kWStride + k] = (_Float16)Wih1[i];
    wh1[r * kWStride + k] = (_Float16)Whh1[i];
  }
  for (int i = tid; i < 160 * 64; i += 256) {
    const int r = i >> 6, k = i & 63;
    whd[r * kWStride + k] = (_Float16)headw[i];
  }
  if (tid < 256) { b0f[tid] = bih0[tid] + bhh0[tid]; b1f[tid] = bih1[tid] + bhh1[tid]; }
  if (tid < 160) bhf[tid] = headb[tid];
  for (int i = tid; i < 512; i += 256) wi0[i] = Wih0[i];

  // ---- stage this wave's x tile: 16 rows x (20*2) f32 ----
  const long long rowbase = (long long)blockIdx.x * kRowsPerBlock + wave * kRowsPerWave;
  float* xbw = (float*)(smem + OFF_X) + wave * kRowsPerWave * 40;
  for (int i = lane; i < kRowsPerWave * 40; i += 32)
    xbw[i] = x[(rowbase + i / 40) * 40 + (i % 40)];

  __syncthreads();

  _Float16* hb = (_Float16*)(smem + OFF_HB) + wave * kRowsPerWave * kWStride;

  v8f z8 = {};
  v8f c0[4], c1[4];
#pragma unroll
  for (int j = 0; j < 4; ++j) { c0[j] = z8; c1[j] = z8; }
  v16h h0A0 = {}, h0A1 = {}, h1A0 = {}, h1A1 = {};

#pragma unroll 1
  for (int t = 0; t < kT; ++t) {
    // per-row input values x[m][t][0..1] for this lane's 8 C/D rows
    float2 xp[8];
#pragma unroll
    for (int r = 0; r < 8; ++r)
      xp[r] = *(const float2*)(xbw + (r + 8 * g) * 40 + t * 2);

    // ================= layer 0 =================
#pragma unroll
    for (int j = 0; j < 4; ++j) {           // hidden tile (16 cols)
      v8f acc[4];
#pragma unroll
      for (int q = 0; q < 4; ++q) {         // gate: i,f,g,o
        const int nt  = q * 4 + j;
        const int col = nt * 16 + n;
        const float2 w01 = *(const float2*)(wi0 + col * 2);
        const float  bb  = b0f[col];
#pragma unroll
        for (int r = 0; r < 8; ++r)
          acc[q][r] = fmaf(w01.x, xp[r].x, fmaf(w01.y, xp[r].y, bb));
        acc[q] = wmma16(h0A0, load_bfrag(wh0, nt, 0, n, g), acc[q]);
        acc[q] = wmma16(h0A1, load_bfrag(wh0, nt, 1, n, g), acc[q]);
      }
#pragma unroll
      for (int r = 0; r < 8; ++r) {
        const float ig = fast_sigmoid(acc[0][r]);
        const float fg = fast_sigmoid(acc[1][r]);
        const float gg = fast_tanh(acc[2][r]);
        const float og = fast_sigmoid(acc[3][r]);
        const float cv = fmaf(fg, c0[j][r], ig * gg);
        c0[j][r] = cv;
        hb[(r + 8 * g) * kWStride + j * 16 + n] = (_Float16)(og * fast_tanh(cv));
      }
    }
    h0A0 = load_afrag(hb, 0, n, g);
    h0A1 = load_afrag(hb, 1, n, g);

    // ================= layer 1 =================
#pragma unroll
    for (int j = 0; j < 4; ++j) {
      v8f acc[4];
#pragma unroll
      for (int q = 0; q < 4; ++q) {
        const int nt = q * 4 + j;
        const float bb = b1f[nt * 16 + n];
#pragma unroll
        for (int r = 0; r < 8; ++r) acc[q][r] = bb;
        acc[q] = wmma16(h0A0, load_bfrag(wi1, nt, 0, n, g), acc[q]);
        acc[q] = wmma16(h0A1, load_bfrag(wi1, nt, 1, n, g), acc[q]);
        acc[q] = wmma16(h1A0, load_bfrag(wh1, nt, 0, n, g), acc[q]);
        acc[q] = wmma16(h1A1, load_bfrag(wh1, nt, 1, n, g), acc[q]);
      }
#pragma unroll
      for (int r = 0; r < 8; ++r) {
        const float ig = fast_sigmoid(acc[0][r]);
        const float fg = fast_sigmoid(acc[1][r]);
        const float gg = fast_tanh(acc[2][r]);
        const float og = fast_sigmoid(acc[3][r]);
        const float cv = fmaf(fg, c1[j][r], ig * gg);
        c1[j][r] = cv;
        hb[(r + 8 * g) * kWStride + j * 16 + n] = (_Float16)(og * fast_tanh(cv));
      }
    }
    h1A0 = load_afrag(hb, 0, n, g);
    h1A1 = load_afrag(hb, 1, n, g);
  }

  // ================= head: out = h1 @ head_w^T + head_b =================
#pragma unroll
  for (int nt = 0; nt < 10; ++nt) {
    const int col = nt * 16 + n;
    const float bb = bhf[col];
    v8f acc;
#pragma unroll
    for (int r = 0; r < 8; ++r) acc[r] = bb;
    acc = wmma16(h1A0, load_bfrag(whd, nt, 0, n, g), acc);
    acc = wmma16(h1A1, load_bfrag(whd, nt, 1, n, g), acc);
#pragma unroll
    for (int r = 0; r < 8; ++r)
      out[(rowbase + r + 8 * g) * 160 + col] = acc[r];
  }
}

extern "C" void kernel_launch(void* const* d_in, const int* in_sizes, int n_in,
                              void* d_out, int out_size, void* d_ws, size_t ws_size,
                              hipStream_t stream) {
  const float* x     = (const float*)d_in[0];
  const float* Wih0  = (const float*)d_in[1];
  const float* Whh0  = (const float*)d_in[2];
  const float* bih0  = (const float*)d_in[3];
  const float* bhh0  = (const float*)d_in[4];
  const float* Wih1  = (const float*)d_in[5];
  const float* Whh1  = (const float*)d_in[6];
  const float* bih1  = (const float*)d_in[7];
  const float* bhh1  = (const float*)d_in[8];
  const float* headw = (const float*)d_in[9];
  const float* headb = (const float*)d_in[10];
  float* out = (float*)d_out;

  const int B = in_sizes[0] / (kT * 2);          // 131072
  const int grid = B / kRowsPerBlock;            // 1024 blocks of 8 waves

  (void)hipFuncSetAttribute((const void*)lstm2_head_wmma_kernel,
                            hipFuncAttributeMaxDynamicSharedMemorySize, SMEM_TOTAL);
  lstm2_head_wmma_kernel<<<grid, 256, SMEM_TOTAL, stream>>>(
      x, Wih0, Whh0, bih0, bhh0, Wih1, Whh1, bih1, bhh1, headw, headb, out);
}
